// LaVoce400_29119878267240
// MI455X (gfx1250) — matrix operand order repaired and checked by reading, not compiled, not run on hardware
//
#include <hip/hip_runtime.h>
#include <hip/hip_bf16.h>
#include <math.h>

// ---------------------------------------------------------------------------
// LaVoce vocoder forward for MI455X (gfx1250, wave32, WMMA).
// All dense algebra is routed through a bf16 WMMA GEMM (v_wmma_f32_16x16x32_bf16);
// the GRU recurrence keeps only h@Whh^T on the critical path, computed from an
// LDS-resident bf16 hidden state. Sample-rate filtering is fused VALU kernels.
// Round-2 tuning: branch-free A-fragment boundary handling (zero-row redirect)
// and dual WMMA accumulators (K unrolled by 64) to break the D->C chain.
// ---------------------------------------------------------------------------

typedef __attribute__((ext_vector_type(16))) __bf16        v16bf;
typedef __attribute__((ext_vector_type(8)))  float         v8f;
typedef __attribute__((ext_vector_type(4)))  unsigned int  u32x4;

union LvFrag { v16bf v; u32x4 q[2]; };

#define LV_GAINA 0.6907755278982137f   // 6 * log(10)/20
#define LV_PI    3.14159265358979323846f

__device__ __forceinline__ unsigned short lv_f2b(float f) {
  unsigned int u = __builtin_bit_cast(unsigned int, f);
  unsigned int r = u + 0x7FFFu + ((u >> 16) & 1u);   // round-to-nearest-even
  return (unsigned short)(r >> 16);
}

__device__ __forceinline__ float lv_act(float v, int act) {
  switch (act) {
    case 1: return tanhf(v);
    case 2: return 1.f / (1.f + expf(-v));
    case 3: return v > 0.f ? v : 0.2f * v;                 // leaky relu 0.2
    case 4: return expf(v);
    case 5: return expf(LV_GAINA * tanhf(v));              // filter gain
    case 6: return expf(-(v > 0.f ? v : 0.f) - LV_GAINA);  // comb cg
    default: return v;
  }
}

// ---------------------------------------------------------------------------
// Generic WMMA GEMM:  Out[M x Npad] = act( A[M x Kpad](bf16) * W^T + bias )
// W stored row-major (Npad x Kpad) bf16 (row n = output column n).
// Optional causal time-shift of A rows (tap of a temporal conv): out-of-range
// rows are redirected to a zeroed scratch row (zrow) so the K-loop stays
// branch-free (no exec save/restore around the loads).
// Fragment layouts per CDNA5 ISA 7.12.2: A row-per-lane with K halves split
// by lane>=16; B column-per-lane, K 8h..8h+7 and 16+8h..16+8h+7 per half-wave, as A.
// ---------------------------------------------------------------------------
__global__ void lv_gemm(const unsigned short* __restrict__ A, int lda,
                        const unsigned short* __restrict__ W, int ldw,
                        const float* __restrict__ bias, int biasN,
                        float* __restrict__ Out, int ldo,
                        int M, int Npad, int Kpad,
                        int act, int accum, int tshift, int Tdim,
                        int rowScale, int rowOff,
                        const unsigned short* __restrict__ zrow) {
  const int wave = threadIdx.x >> 5;
  const int lane = threadIdx.x & 31;
  const int tilesN = Npad >> 4;
  long tile = (long)blockIdx.x * 8 + wave;
  if (tile >= (long)(M >> 4) * tilesN) return;   // wave-uniform
  const int tn = (int)(tile % tilesN);
  const int tm = (int)(tile / tilesN);
  const int half = lane >> 4;
  const int lr = lane & 15;

  int m = tm * 16 + lr;
  long arow = m;
  bool valid = true;
  if (Tdim > 0) {                     // causal-conv tap: shift within time axis
    int b = m / Tdim;
    int t = m - b * Tdim + tshift;
    valid = (t >= 0) && (t < Tdim);
    arow = (long)b * Tdim + (valid ? t : 0);
  }
  const unsigned short* aptr =
      valid ? (A + arow * (long)lda + half * 8) : (zrow + half * 8);
  const unsigned short* wptr = W + (long)(tn * 16 + lr) * ldw + half * 8;

  v8f acc0 = {}, acc1 = {};
  int k = 0;
  for (; k + 64 <= Kpad; k += 64) {
    LvFrag fa0, fb0, fa1, fb1;
    fa0.q[0] = *(const u32x4*)(aptr + k);
    fa0.q[1] = *(const u32x4*)(aptr + k + 16);
    fb0.q[0] = *(const u32x4*)(wptr + k);
    fb0.q[1] = *(const u32x4*)(wptr + k + 16);
    fa1.q[0] = *(const u32x4*)(aptr + k + 32);
    fa1.q[1] = *(const u32x4*)(aptr + k + 48);
    fb1.q[0] = *(const u32x4*)(wptr + k + 32);
    fb1.q[1] = *(const u32x4*)(wptr + k + 48);
    acc0 = __builtin_amdgcn_wmma_f32_16x16x32_bf16(false, fa0.v, false, fb0.v,
                                                   (short)0, acc0, false, false);
    acc1 = __builtin_amdgcn_wmma_f32_16x16x32_bf16(false, fa1.v, false, fb1.v,
                                                   (short)0, acc1, false, false);
  }
  if (k < Kpad) {                     // odd 32-wide tail (Kpad = 96, 288, ...)
    LvFrag fa, fb;
    fa.q[0] = *(const u32x4*)(aptr + k);
    fa.q[1] = *(const u32x4*)(aptr + k + 16);
    fb.q[0] = *(const u32x4*)(wptr + k);
    fb.q[1] = *(const u32x4*)(wptr + k + 16);
    acc0 = __builtin_amdgcn_wmma_f32_16x16x32_bf16(false, fa.v, false, fb.v,
                                                   (short)0, acc0, false, false);
  }

  const int n = tn * 16 + lr;
  float bv = (bias != nullptr && n < biasN) ? bias[n] : 0.f;
#pragma unroll
  for (int r = 0; r < 8; ++r) {
    int mrow = tm * 16 + half * 8 + r;
    long oidx = ((long)mrow * rowScale + rowOff) * (long)ldo + n;
    float v = acc0[r] + acc1[r] + bv;
    if (accum) v += Out[oidx];
    Out[oidx] = lv_act(v, act);
  }
}

// ---------------------------------------------------------------------------
// GRU scan: gi (input projection + bih) is precomputed by lv_gemm; only the
// recurrent gh = h @ Whh^T stays sequential. 2 blocks x 16 batches each; the
// bf16 hidden state lives in LDS and feeds the WMMA A operand via ds loads.
// ---------------------------------------------------------------------------
__global__ void lv_gru_scan(const unsigned short* __restrict__ whh,
                            const float* __restrict__ gi,
                            const float* __restrict__ bhh,
                            float* __restrict__ cff) {
  extern __shared__ char smem[];
  float*          hs = (float*)smem;                              // 16x256 f32
  unsigned short* hb = (unsigned short*)(smem + 16 * 256 * 4);    // 16x256 bf16
  float*          gh = (float*)(smem + 16 * 256 * 4 + 16 * 256 * 2); // 16x768

  const int bbase = blockIdx.x * 16;
  const int tid = threadIdx.x;
  const int wave = tid >> 5, lane = tid & 31;
  const int half = lane >> 4, lr = lane & 15;

  for (int i = tid; i < 16 * 256; i += 256) { hs[i] = 0.f; hb[i] = 0; }
  __syncthreads();

  for (int t = 0; t < 2000; ++t) {
    // gh = h(16x256) @ Whh^T(768x256): 48 N-tiles, 6 per wave, K in 8 wmma steps
    for (int i2 = 0; i2 < 6; ++i2) {
      int tn = wave * 6 + i2;
      const unsigned short* aptr = hb + lr * 256 + half * 8;
      const unsigned short* wptr = whh + (long)(tn * 16 + lr) * 256 + half * 8;
      v8f acc0 = {}, acc1 = {};
      for (int k = 0; k < 256; k += 64) {
        LvFrag fa0, fb0, fa1, fb1;
        fa0.q[0] = *(const u32x4*)(aptr + k);        // ds_load_b128 from LDS
        fa0.q[1] = *(const u32x4*)(aptr + k + 16);
        fb0.q[0] = *(const u32x4*)(wptr + k);
        fb0.q[1] = *(const u32x4*)(wptr + k + 16);
        fa1.q[0] = *(const u32x4*)(aptr + k + 32);
        fa1.q[1] = *(const u32x4*)(aptr + k + 48);
        fb1.q[0] = *(const u32x4*)(wptr + k + 32);
        fb1.q[1] = *(const u32x4*)(wptr + k + 48);
        acc0 = __builtin_amdgcn_wmma_f32_16x16x32_bf16(false, fa0.v, false, fb0.v,
                                                       (short)0, acc0, false, false);
        acc1 = __builtin_amdgcn_wmma_f32_16x16x32_bf16(false, fa1.v, false, fb1.v,
                                                       (short)0, acc1, false, false);
      }
      int n = tn * 16 + lr;
#pragma unroll
      for (int r = 0; r < 8; ++r)
        gh[(half * 8 + r) * 768 + n] = acc0[r] + acc1[r];
    }
    __syncthreads();

    // gate math over 16x256 hidden units
    for (int i = tid; i < 16 * 256; i += 256) {
      int lb = i >> 8, j = i & 255;
      int b = bbase + lb;
      long grow = ((long)b * 2000 + t) * 768;
      float xr = gi[grow + j], xz = gi[grow + 256 + j], xn = gi[grow + 512 + j];
      float hr = gh[lb * 768 + j] + bhh[j];
      float hz = gh[lb * 768 + 256 + j] + bhh[256 + j];
      float hn = gh[lb * 768 + 512 + j] + bhh[512 + j];
      float rg = 1.f / (1.f + expf(-(xr + hr)));
      float zg = 1.f / (1.f + expf(-(xz + hz)));
      float ng = tanhf(xn + rg * hn);
      float h = (1.f - zg) * ng + zg * hs[i];
      hs[i] = h;
      hb[i] = lv_f2b(h);
      cff[((long)b * 2000 + t) * 256 + j] = h;
    }
    __syncthreads();
  }
}

// --------------------------- data-prep kernels ------------------------------

__global__ void lv_zero16(unsigned short* __restrict__ p, int n) {
  int i = blockIdx.x * 256 + threadIdx.x;
  if (i < n) p[i] = 0;
}

// weight f32 -> bf16, generic strides, zero-padded to (Npad x Kpad)
__global__ void lv_w2bf(const float* __restrict__ src, unsigned short* __restrict__ dst,
                        int N, int Kreal, int Npad, int Kpad, int sn, int sk, int off) {
  int i = blockIdx.x * 256 + threadIdx.x;
  if (i >= Npad * Kpad) return;
  int n = i / Kpad, k = i - n * Kpad;
  float v = (n < N && k < Kreal) ? src[(long)n * sn + (long)k * sk + off] : 0.f;
  dst[i] = lv_f2b(v);
}

// activation f32 (ld=lds, first Kreal cols) -> bf16 (ld=ldd, zero-padded)
__global__ void lv_a2bf(const float* __restrict__ src, int lds,
                        unsigned short* __restrict__ dst, int ldd, long M, int Kreal) {
  long i = (long)blockIdx.x * 256 + threadIdx.x;
  if (i >= M * ldd) return;
  long row = i / ldd; int col = (int)(i - row * ldd);
  float v = (col < Kreal) ? src[row * (long)lds + col] : 0.f;
  dst[i] = lv_f2b(v);
}

// concat(features, emb[periods]) -> bf16 (16000 x 96, 84 real)
__global__ void lv_embed(const float* __restrict__ feats, const int* __restrict__ per,
                         const float* __restrict__ emb, unsigned short* __restrict__ X0) {
  long i = (long)blockIdx.x * 256 + threadIdx.x;
  if (i >= 16000L * 96) return;
  int row = (int)(i / 96), c = (int)(i - (long)row * 96);
  float v = 0.f;
  if (c < 20)       v = feats[(long)row * 20 + c];
  else if (c < 84)  v = emb[(long)per[row] * 64 + (c - 20)];
  X0[i] = lv_f2b(v);
}

// per-batch serial prefix of instantaneous frequency
__global__ void lv_phase(const int* __restrict__ per, float* __restrict__ phase0,
                         float* __restrict__ freqb) {
  int b = threadIdx.x;
  if (b >= 32) return;
  float ph = 0.f;
  for (int f = 0; f < 2000; ++f) {
    float fr = 2.f * LV_PI / (float)per[b * 500 + (f >> 2)];
    phase0[b * 2000 + f] = 40.f * ph;   // FS * (cumsum - freq)
    freqb[b * 2000 + f] = fr;
    ph += fr;
  }
}

__global__ void lv_refgen(const float* __restrict__ phase0, const float* __restrict__ freqb,
                          float* __restrict__ xref) {
  long i = (long)blockIdx.x * 256 + threadIdx.x;
  if (i >= 32L * 80000) return;
  int b = (int)(i / 80000); int g = (int)(i - (long)b * 80000);
  int f = g / 40, s = g - f * 40;
  float fr = freqb[b * 2000 + f];
  float arg = fr * (float)(s + 1) + phase0[b * 2000 + f];
  xref[((long)b * 2) * 80000 + g]     = tanhf(sinf(arg));
  xref[((long)b * 2 + 1) * 80000 + g] = tanhf(cosf(arg));
}

// per-output-channel L2 normalisation of generated kernels (+ gain)
__global__ void lv_kern_norm(float* __restrict__ kern, int ldk, int realN,
                             const float* __restrict__ gains, int ldg, int oc, int icK) {
  int row = blockIdx.x, tid = threadIdx.x;
  __shared__ float invs[4];
  if (tid < oc) {
    float s = 0.f;
    for (int i = 0; i < icK; ++i) {
      float v = kern[(long)row * ldk + tid * icK + i];
      s += v * v;
    }
    float g = gains ? gains[(long)row * ldg + tid] : 1.f;
    invs[tid] = g / (1e-6f + sqrtf(s));
  }
  __syncthreads();
  for (int i = tid; i < realN; i += blockDim.x)
    kern[(long)row * ldk + i] *= invs[i / icK];
}

// adaptive FIR over a frame window: 15 taps, S = FS + OL_CONV = 80
__global__ void lv_adconv(const float* __restrict__ kern, int ldk,
                          const float* __restrict__ x0, long bs0,
                          const float* __restrict__ x1, long bs1,
                          float* __restrict__ nb, int ic, int oc) {
  int row = blockIdx.x;
  int b = row / 2000, f = row - b * 2000;
  int tid = threadIdx.x;
  if (tid >= oc * 80) return;
  int o = tid / 80, s = tid - o * 80;
  float acc = 0.f;
  for (int i = 0; i < ic; ++i) {
    const float* xp = (i == 0) ? (x0 + (long)b * bs0) : (x1 + (long)b * bs1);
    const float* kp = kern + (long)row * ldk + (o * ic + i) * 15;
    int gbase = f * 40 + s - 14;
#pragma unroll
    for (int k = 0; k < 15; ++k) {
      int g = gbase + k;
      float xv = (g >= 0 && g < 80000) ? xp[g] : 0.f;
      acc += kp[k] * xv;
    }
  }
  nb[(((long)b * oc + o) * 2000 + f) * 80 + s] = acc;
}

// windowed overlap-add of per-frame outputs (stride-80 staging buffer)
__global__ void lv_oadd(const float* __restrict__ nb, float* __restrict__ out,
                        int oc, int ol) {
  long i = (long)blockIdx.x * 256 + threadIdx.x;
  if (i >= 32L * oc * 80000) return;
  long bo = i / 80000; int g = (int)(i - bo * 80000);
  int f = g / 40, j = g - f * 40;
  const float* nbp = nb + bo * 2000L * 80;
  float v;
  if (j < ol) {
    float w2 = 0.5f + 0.5f * cosf(((float)j + 0.5f) * LV_PI / (float)ol);
    float w1 = 0.5f + 0.5f * cosf(((float)(ol - 1 - j) + 0.5f) * LV_PI / (float)ol);
    float tail = (f > 0) ? nbp[(long)(f - 1) * 80 + 40 + j] : 0.f;
    v = nbp[(long)f * 80 + j] * w1 + tail * w2;
  } else {
    v = nbp[(long)f * 80 + j];
  }
  out[i] = v;
}

// pitch comb filter: 15-tap around lag, S = FS + OL_COMB = 60
__global__ void lv_comb(const float* __restrict__ kern, const float* __restrict__ cg,
                        const float* __restrict__ gg, int ldg,
                        const int* __restrict__ per, const float* __restrict__ x,
                        float* __restrict__ nb) {
  int row = blockIdx.x;
  int b = row / 2000, f = row - b * 2000;
  int s = threadIdx.x;
  if (s >= 60) return;
  int lag = per[b * 500 + (f >> 2)];
  const float* xb = x + (long)b * 80000;
  float conv = 0.f;
  int gbase = f * 40 + s - lag - 7;   // LP = 7
#pragma unroll
  for (int k = 0; k < 15; ++k) {
    int g = gbase + k;
    conv += kern[(long)row * 16 + k] * ((g >= 0 && g < 80000) ? xb[g] : 0.f);
  }
  int gd = f * 40 + s;
  float direct = (gd < 80000) ? xb[gd] : 0.f;
  nb[(long)row * 80 + s] = gg[(long)row * ldg] * (direct + cg[(long)row * ldg] * conv);
}

// temporal envelope (log of 4-sample mean |x|) + DC split
__global__ void lv_env(const float* __restrict__ x, long bs, float* __restrict__ tenv) {
  int i = blockIdx.x * blockDim.x + threadIdx.x;
  if (i >= 64000) return;
  int b = i / 2000, f = i - b * 2000;
  const float* xb = x + (long)b * bs + f * 40;
  float e[10]; float avg = 0.f;
#pragma unroll
  for (int j = 0; j < 10; ++j) {
    float s = 0.f;
    for (int q = 0; q < 4; ++q) s += fabsf(xb[j * 4 + q]);
    e[j] = logf(s * 0.25f + 1.52587890625e-05f);   // + 2^-16
    avg += e[j];
  }
  avg *= 0.1f;
  for (int j = 0; j < 10; ++j) tenv[(long)i * 11 + j] = e[j] - avg;
  tenv[(long)i * 11 + 10] = avg;
}

// concat(cf, tenv) -> bf16 (64000 x 288, 267 real)
__global__ void lv_buildf(const float* __restrict__ cf, const float* __restrict__ tenv,
                          unsigned short* __restrict__ fbuf) {
  long i = (long)blockIdx.x * 256 + threadIdx.x;
  if (i >= 64000L * 288) return;
  long row = i / 288; int c = (int)(i - row * 288);
  float v = 0.f;
  if (c < 256)      v = cf[row * 256 + c];
  else if (c < 267) v = tenv[row * 11 + (c - 256)];
  fbuf[i] = lv_f2b(v);
}

// td-shaper application: y = alpha*x (+ inno_alpha*inno_x)
__global__ void lv_tdmul(const float* __restrict__ alpha, int lda,
                         const float* __restrict__ xin, long bsx,
                         const float* __restrict__ alpha2, const float* __restrict__ ix,
                         float* __restrict__ y) {
  long i = (long)blockIdx.x * 256 + threadIdx.x;
  if (i >= 32L * 80000) return;
  int b = (int)(i / 80000); int g = (int)(i - (long)b * 80000);
  int f = g / 40, s = g - f * 40;
  long r = ((long)b * 2000 + f) * lda + s;
  float v = alpha[r] * xin[(long)b * bsx + g];
  if (alpha2) v += alpha2[r] * ix[r];
  y[i] = v;
}

__global__ void lv_nsmul(const float* __restrict__ alpha, const float* __restrict__ noise,
                         float* __restrict__ nz) {
  long i = (long)blockIdx.x * 256 + threadIdx.x;
  if (i >= 32L * 80000) return;
  int b = (int)(i / 80000); int g = (int)(i - (long)b * 80000);
  int f = g / 40, s = g - f * 40;
  nz[i] = alpha[((long)b * 2000 + f) * 48 + s] * noise[((long)b * 2000 + f) * 40 + s];
}

// ---------------------------------------------------------------------------
// host orchestration
// ---------------------------------------------------------------------------
extern "C" void kernel_launch(void* const* d_in, const int* in_sizes, int n_in,
                              void* d_out, int out_size, void* d_ws, size_t ws_size,
                              hipStream_t stream) {
  (void)in_sizes; (void)out_size; (void)ws_size;
  if (n_in < 84 || !d_out || !d_ws) return;

  const float* features = (const float*)d_in[0];
  const int*   periods  = (const int*)d_in[1];
  const float* noise    = (const float*)d_in[2];
  const float* embw     = (const float*)d_in[3];
#define P(i) ((const float*)d_in[(i)])

  char* wsp = (char*)d_ws;
  size_t off = 0;
  auto alloc = [&](size_t bytes) -> void* {
    void* p = wsp + off;
    off = (off + bytes + 255) & ~(size_t)255;
    return p;
  };

  // ---- zero row for branch-free boundary taps (max Kpad = 288) ----
  unsigned short* zrow = (unsigned short*)alloc(512 * 2);
  lv_zero16<<<2, 256, 0, stream>>>(zrow, 512);

  // ---- bf16 weights (zero-padded) ----
  unsigned short* wc1  = (unsigned short*)alloc(3L * 256 * 96 * 2);
  unsigned short* wc2  = (unsigned short*)alloc(3L * 256 * 256 * 2);
  unsigned short* wtc  = (unsigned short*)alloc(4L * 256 * 256 * 2);
  unsigned short* wih  = (unsigned short*)alloc(768L * 256 * 2);
  unsigned short* whhB = (unsigned short*)alloc(768L * 256 * 2);
  unsigned short* wns1 = (unsigned short*)alloc(2L * 48 * 256 * 2);
  unsigned short* wns2 = (unsigned short*)alloc(2L * 48 * 64 * 2);

  auto w2bf = [&](const float* src, unsigned short* dst, int N, int Kreal,
                  int Npad, int Kpad, int sn, int sk, int ntaps) {
    int blocks = (Npad * Kpad + 255) / 256;
    for (int j = 0; j < ntaps; ++j)
      lv_w2bf<<<blocks, 256, 0, stream>>>(src, dst + (long)j * Npad * Kpad,
                                          N, Kreal, Npad, Kpad, sn, sk, j);
  };

  w2bf(P(4),  wc1, 256, 84, 256, 96, 252, 3, 3);       // (D,84,3)
  w2bf(P(6),  wc2, 256, 256, 256, 256, 768, 3, 3);     // (D,D,3)
  w2bf(P(8),  wtc, 256, 256, 256, 256, 4, 1024, 4);    // (c,d,k): n=d, kk=c
  w2bf(P(10), wih, 768, 256, 768, 256, 256, 1, 1);
  w2bf(P(11), whhB, 768, 256, 768, 256, 256, 1, 1);
  w2bf(P(14), wns1, 40, 256, 48, 256, 512, 2, 2);
  w2bf(P(16), wns2, 40, 40, 48, 64, 80, 2, 2);

  struct ACDesc { int ic, oc, Npad, base; unsigned short *ck, *fg; };
  ACDesc ac[6] = {
    {2,1,32,18,nullptr,nullptr},  // afp
    {2,2,64,22,nullptr,nullptr},  // afm
    {1,2,32,26,nullptr,nullptr},  // af1
    {2,2,64,30,nullptr,nullptr},  // af2
    {2,1,32,34,nullptr,nullptr},  // af3
    {2,1,32,38,nullptr,nullptr},  // af4
  };
  for (int i = 0; i < 6; ++i) {
    ac[i].ck = (unsigned short*)alloc((size_t)ac[i].Npad * 256 * 2);
    ac[i].fg = (unsigned short*)alloc(16L * 256 * 2);
    w2bf(P(ac[i].base),     ac[i].ck, ac[i].ic * ac[i].oc * 15, 256, ac[i].Npad, 256, 256, 1, 1);
    w2bf(P(ac[i].base + 2), ac[i].fg, ac[i].oc, 256, 16, 256, 256, 1, 1);
  }

  unsigned short *wcck[2], *wcfg[2], *wcgg[2];
  for (int i = 0; i < 2; ++i) {
    int base = 42 + 6 * i;
    wcck[i] = (unsigned short*)alloc(16L * 256 * 2);
    wcfg[i] = (unsigned short*)alloc(16L * 256 * 2);
    wcgg[i] = (unsigned short*)alloc(16L * 256 * 2);
    w2bf(P(base),     wcck[i], 15, 256, 16, 256, 256, 1, 1);
    w2bf(P(base + 2), wcfg[i], 1, 256, 16, 256, 256, 1, 1);
    w2bf(P(base + 4), wcgg[i], 1, 256, 16, 256, 256, 1, 1);
  }

  auto tdw1 = [&](int idx) -> unsigned short* {
    unsigned short* p = (unsigned short*)alloc(2L * 48 * 288 * 2);
    w2bf(P(idx), p, 40, 267, 48, 288, 534, 2, 2);
    return p;
  };
  auto tdw2 = [&](int idx) -> unsigned short* {
    unsigned short* p = (unsigned short*)alloc(2L * 48 * 64 * 2);
    w2bf(P(idx), p, 40, 40, 48, 64, 80, 2, 2);
    return p;
  };
  unsigned short* wt1a1 = tdw1(54); unsigned short* wt1a2 = tdw2(56);
  unsigned short* wt1b1 = tdw1(58); unsigned short* wt1b2 = tdw2(60);
  unsigned short* wt1c1 = tdw1(62); unsigned short* wt1c2 = tdw2(64);
  unsigned short* wt2a1 = tdw1(66); unsigned short* wt2a2 = tdw2(68);
  unsigned short* wt3a1 = tdw1(70); unsigned short* wt3a2 = tdw2(72);

  unsigned short* wpost[5];  // post_cf1, post_cf2, post_af1, post_af2, post_af3
  for (int i = 0; i < 5; ++i) {
    wpost[i] = (unsigned short*)alloc(2L * 256 * 256 * 2);
    w2bf(P(74 + 2 * i), wpost[i], 256, 256, 256, 256, 512, 2, 2);
  }

  // ---- activations ----
  unsigned short* X0  = (unsigned short*)alloc(16000L * 96 * 2);
  float* t1f = (float*)alloc(16000L * 256 * 4);
  unsigned short* t1b = (unsigned short*)alloc(16000L * 256 * 2);
  float* t2f = (float*)alloc(16000L * 256 * 4);
  unsigned short* t2b = (unsigned short*)alloc(16000L * 256 * 2);
  float* c4f = (float*)alloc(64000L * 256 * 4);
  unsigned short* c4b = (unsigned short*)alloc(64000L * 256 * 2);
  float* gi  = (float*)alloc(64000L * 768 * 4);
  float* cfA = (float*)alloc(64000L * 256 * 4);
  float* cfB = (float*)alloc(64000L * 256 * 4);
  unsigned short* cfb = (unsigned short*)alloc(64000L * 256 * 2);
  float* phase0 = (float*)alloc(32L * 2000 * 4);
  float* freqb  = (float*)alloc(32L * 2000 * 4);
  float* xref   = (float*)alloc(32L * 2 * 80000 * 4);
  float* kern   = (float*)alloc(64000L * 64 * 4);
  float* fgo    = (float*)alloc(64000L * 16 * 4);
  float* cgo    = (float*)alloc(64000L * 16 * 4);
  float* ggo    = (float*)alloc(64000L * 16 * 4);
  float* newbuf = (float*)alloc(32L * 2 * 2000 * 80 * 4);
  float* ybufA  = (float*)alloc(32L * 2 * 80000 * 4);
  float* ybufB  = (float*)alloc(32L * 2 * 80000 * 4);
  float* xa     = (float*)alloc(32L * 80000 * 4);
  float* nz     = (float*)alloc(32L * 80000 * 4);
  float* y2     = (float*)alloc(32L * 80000 * 4);
  float* yc     = (float*)alloc(32L * 80000 * 4);
  float* tenv   = (float*)alloc(64000L * 11 * 4);
  unsigned short* fbuf = (unsigned short*)alloc(64000L * 288 * 2);
  float* a1o    = (float*)alloc(64000L * 48 * 4);
  unsigned short* a1bb = (unsigned short*)alloc(64000L * 64 * 2);
  float* alpha  = (float*)alloc(64000L * 48 * 4);
  float* alpha2 = (float*)alloc(64000L * 48 * 4);
  float* ixo    = (float*)alloc(64000L * 48 * 4);

  // ---- helpers ----
  auto gemm = [&](const unsigned short* A, int lda, const unsigned short* W, int ldw,
                  const float* bias, int biasN, float* Out, int ldo,
                  int M, int Npad, int Kpad, int act, int accum,
                  int tshift, int Tdim, int rowScale, int rowOff) {
    long ntiles = (long)(M >> 4) * (Npad >> 4);
    int blocks = (int)((ntiles + 7) / 8);
    lv_gemm<<<blocks, 256, 0, stream>>>(A, lda, W, ldw, bias, biasN, Out, ldo,
                                        M, Npad, Kpad, act, accum, tshift, Tdim,
                                        rowScale, rowOff, zrow);
  };
  auto cconv2 = [&](const unsigned short* A, int lda, unsigned short* Wb, int ldw,
                    long tapStride, const float* bias, int biasN, float* Out, int ldo,
                    int M, int Npad, int Kpad, int act, int Tdim) {
    gemm(A, lda, Wb, ldw, bias, biasN, Out, ldo, M, Npad, Kpad, 0, 0, -1, Tdim, 1, 0);
    gemm(A, lda, Wb + tapStride, ldw, nullptr, 0, Out, ldo, M, Npad, Kpad, act, 1, 0, Tdim, 1, 0);
  };
  auto a2bf = [&](const float* src, int lds, unsigned short* dst, int ldd,
                  long M, int Kreal) {
    long tot = M * ldd;
    lv_a2bf<<<(int)((tot + 255) / 256), 256, 0, stream>>>(src, lds, dst, ldd, M, Kreal);
  };

  float* cf_cur = cfA; float* cf_nxt = cfB;

  auto ftrans = [&](unsigned short* w, const float* b) {
    cconv2(cfb, 256, w, 256, 65536L, b, 256, cf_nxt, 256, 64000, 256, 256, 1, 2000);
    float* t = cf_cur; cf_cur = cf_nxt; cf_nxt = t;
    a2bf(cf_cur, 256, cfb, 256, 64000, 256);
  };

  auto adconv = [&](ACDesc& a, const float* x0p, long bs0, const float* x1p, long bs1,
                    float* outp) {
    int realN = a.ic * a.oc * 15;
    gemm(cfb, 256, a.ck, 256, P(a.base + 1), realN, kern, a.Npad, 64000, a.Npad, 256, 0, 0, 0, 0, 1, 0);
    gemm(cfb, 256, a.fg, 256, P(a.base + 3), a.oc, fgo, 16, 64000, 16, 256, 5, 0, 0, 0, 1, 0);
    lv_kern_norm<<<64000, 64, 0, stream>>>(kern, a.Npad, realN, fgo, 16, a.oc, a.ic * 15);
    lv_adconv<<<64000, 256, 0, stream>>>(kern, a.Npad, x0p, bs0, x1p, bs1, newbuf, a.ic, a.oc);
    long tot = 32L * a.oc * 80000;
    lv_oadd<<<(int)((tot + 255) / 256), 256, 0, stream>>>(newbuf, outp, a.oc, 40);
  };

  auto comb = [&](int ci, const float* xin, float* outp) {
    int base = 42 + 6 * ci;
    gemm(cfb, 256, wcck[ci], 256, P(base + 1), 15, kern, 16, 64000, 16, 256, 0, 0, 0, 0, 1, 0);
    gemm(cfb, 256, wcfg[ci], 256, P(base + 3), 1, cgo, 16, 64000, 16, 256, 6, 0, 0, 0, 1, 0);
    gemm(cfb, 256, wcgg[ci], 256, P(base + 5), 1, ggo, 16, 64000, 16, 256, 5, 0, 0, 0, 1, 0);
    lv_kern_norm<<<64000, 64, 0, stream>>>(kern, 16, 15, nullptr, 16, 1, 15);
    lv_comb<<<64000, 64, 0, stream>>>(kern, cgo, ggo, 16, periods, xin, newbuf);
    long tot = 32L * 80000;
    lv_oadd<<<(int)((tot + 255) / 256), 256, 0, stream>>>(newbuf, outp, 1, 20);
  };

  auto tdshape = [&](unsigned short* a1w, const float* a1b, unsigned short* a2w, const float* a2b_,
                     unsigned short* b1w, const float* b1b, unsigned short* b2w, const float* b2b_,
                     unsigned short* c1w, const float* c1b, unsigned short* c2w, const float* c2b_,
                     int innovate, const float* xin, long bsx, float* yout) {
    lv_env<<<250, 256, 0, stream>>>(xin, bsx, tenv);
    { long tot = 64000L * 288;
      lv_buildf<<<(int)((tot + 255) / 256), 256, 0, stream>>>(cf_cur, tenv, fbuf); }
    cconv2(fbuf, 288, a1w, 288, 48L * 288, a1b, 40, a1o, 48, 64000, 48, 288, 3, 2000);
    a2bf(a1o, 48, a1bb, 64, 64000, 40);
    cconv2(a1bb, 64, a2w, 64, 48L * 64, a2b_, 40, alpha, 48, 64000, 48, 64, 4, 2000);
    if (innovate) {
      cconv2(fbuf, 288, b1w, 288, 48L * 288, b1b, 40, a1o, 48, 64000, 48, 288, 3, 2000);
      a2bf(a1o, 48, a1bb, 64, 64000, 40);
      cconv2(a1bb, 64, b2w, 64, 48L * 64, b2b_, 40, alpha2, 48, 64000, 48, 64, 4, 2000);
      cconv2(fbuf, 288, c1w, 288, 48L * 288, c1b, 40, a1o, 48, 64000, 48, 288, 3, 2000);
      a2bf(a1o, 48, a1bb, 64, 64000, 40);
      cconv2(a1bb, 64, c2w, 64, 48L * 64, c2b_, 40, ixo, 48, 64000, 48, 64, 1, 2000);
    }
    long tot = 32L * 80000;
    lv_tdmul<<<(int)((tot + 255) / 256), 256, 0, stream>>>(alpha, 48, xin, bsx,
                                                           innovate ? alpha2 : nullptr, ixo, yout);
  };

  // ================= feature net =================
  { long tot = 16000L * 96;
    lv_embed<<<(int)((tot + 255) / 256), 256, 0, stream>>>(features, periods, embw, X0); }
  // conv1 (k=3, pad(1,1)): taps t-1, t, t+1
  gemm(X0, 96, wc1, 96, P(5), 256, t1f, 256, 16000, 256, 96, 0, 0, -1, 500, 1, 0);
  gemm(X0, 96, wc1 + 256 * 96, 96, nullptr, 0, t1f, 256, 16000, 256, 96, 0, 1, 0, 500, 1, 0);
  gemm(X0, 96, wc1 + 2 * 256 * 96, 96, nullptr, 0, t1f, 256, 16000, 256, 96, 1, 1, 1, 500, 1, 0);
  a2bf(t1f, 256, t1b, 256, 16000, 256);
  // conv2 (k=3, pad(2,0)): taps t-2, t-1, t
  gemm(t1b, 256, wc2, 256, P(7), 256, t2f, 256, 16000, 256, 256, 0, 0, -2, 500, 1, 0);
  gemm(t1b, 256, wc2 + 65536, 256, nullptr, 0, t2f, 256, 16000, 256, 256, 0, 1, -1, 500, 1, 0);
  gemm(t1b, 256, wc2 + 131072, 256, nullptr, 0, t2f, 256, 16000, 256, 256, 1, 1, 0, 500, 1, 0);
  a2bf(t2f, 256, t2b, 256, 16000, 256);
  // 4x temporal upsample (einsum btc,cdk->btkd) with interleaved output rows
  for (int ks = 0; ks < 4; ++ks)
    gemm(t2b, 256, wtc + ks * 65536, 256, P(9), 256, c4f, 256, 16000, 256, 256, 1, 0, 0, 0, 4, ks);
  a2bf(c4f, 256, c4b, 256, 64000, 256);
  // GRU input projection (parallel) then recurrent scan (WMMA from LDS)
  gemm(c4b, 256, wih, 256, P(12), 768, gi, 768, 64000, 768, 256, 0, 0, 0, 0, 1, 0);
  lv_gru_scan<<<2, 256, 16 * 256 * 4 + 16 * 256 * 2 + 16 * 768 * 4, stream>>>(whhB, gi, P(13), cfA);
  a2bf(cf_cur, 256, cfb, 256, 64000, 256);

  // ================= reference excitation =================
  lv_phase<<<1, 32, 0, stream>>>(periods, phase0, freqb);
  { long tot = 32L * 80000;
    lv_refgen<<<(int)((tot + 255) / 256), 256, 0, stream>>>(phase0, freqb, xref); }

  // ================= synthesis chain =================
  adconv(ac[0], xref, 160000, xref + 80000, 160000, xa);                 // afp
  // noise shaper (uses pre-transform cf)
  cconv2(cfb, 256, wns1, 256, 48L * 256, P(15), 40, a1o, 48, 64000, 48, 256, 3, 2000);
  a2bf(a1o, 48, a1bb, 64, 64000, 40);
  cconv2(a1bb, 64, wns2, 64, 48L * 64, P(17), 40, alpha, 48, 64000, 48, 64, 4, 2000);
  { long tot = 32L * 80000;
    lv_nsmul<<<(int)((tot + 255) / 256), 256, 0, stream>>>(alpha, noise, nz); }
  adconv(ac[1], xa, 80000, nz, 80000, ybufA);                            // afm
  tdshape(wt1a1, P(55), wt1a2, P(57), wt1b1, P(59), wt1b2, P(61),
          wt1c1, P(63), wt1c2, P(65), 1, ybufA + 80000, 160000, y2);     // td1
  adconv(ac[3], ybufA, 160000, y2, 80000, ybufB);                        // af2
  ftrans(wpost[3], P(81));                                               // post_af2
  tdshape(wt2a1, P(67), wt2a2, P(69), nullptr, nullptr, nullptr, nullptr,
          nullptr, nullptr, nullptr, nullptr, 0, ybufB + 80000, 160000, y2); // td2
  adconv(ac[4], ybufB, 160000, y2, 80000, yc);                           // af3
  ftrans(wpost[4], P(83));                                               // post_af3
  comb(0, yc, xa);                                                       // cf1
  ftrans(wpost[0], P(75));                                               // post_cf1
  comb(1, xa, nz);                                                       // cf2
  ftrans(wpost[1], P(77));                                               // post_cf2
  adconv(ac[2], nz, 80000, nullptr, 0, ybufA);                           // af1 (ic=1)
  ftrans(wpost[2], P(79));                                               // post_af1
  tdshape(wt3a1, P(71), wt3a2, P(73), nullptr, nullptr, nullptr, nullptr,
          nullptr, nullptr, nullptr, nullptr, 0, ybufA + 80000, 160000, y2); // td3
  adconv(ac[5], ybufA, 160000, y2, 80000, (float*)d_out);                // af4 -> output
#undef P
}
